// NeuralPromptProducerLayer_65111704207696
// MI455X (gfx1250) — matrix operand order repaired and checked by reading, not compiled, not run on hardware
//
#include <hip/hip_runtime.h>
#include <hip/hip_bf16.h>
#include <hip/hip_fp16.h>

typedef __attribute__((ext_vector_type(16))) _Float16 v16h;
typedef __attribute__((ext_vector_type(8)))  _Float16 v8h;
typedef __attribute__((ext_vector_type(4)))  _Float16 v4h;
typedef __attribute__((ext_vector_type(8)))  float    v8f;
typedef __attribute__((ext_vector_type(4)))  unsigned int u32x4;
typedef __attribute__((ext_vector_type(8)))  int      i32x8;
typedef __attribute__((ext_vector_type(4)))  int      i32x4;

#define cB   8
#define cT   704
#define cD   1024
#define cR   16
#define cKT  4
#define cHD  64
#define cSD  32
#define cE   3
#define BT   (cB*cT)
#define NKT  (cT/64)   // 11 key tiles of 64

// ---------------- async global->LDS (CDNA5), ASYNCcnt-tracked ----------------
__device__ __forceinline__ void async_g2l_b128(const v8h* g, v8h* l) {
#if defined(__gfx1250__)
  // VDST = per-lane LDS byte address, VADDR = 64-bit global address (GV mode).
  asm volatile("global_load_async_to_lds_b128 %0, %1, off"
               :: "v"((unsigned)(size_t)l), "v"(g) : "memory");
#else
  *l = *g;
#endif
}
template<int N>
__device__ __forceinline__ void wait_async() {
#if defined(__gfx1250__)
  asm volatile("s_wait_asynccnt %0" :: "i"(N) : "memory");
#endif
}

// ---------------- WMMA fragment loaders (CDNA5 16-bit layouts), 128-bit LDS paths ----------------
__device__ __forceinline__ v16h pack2(v8h lo, v8h hi) {
  union { v8h h8[2]; v16h v; } u;
  u.h8[0] = lo; u.h8[1] = hi;
  return u.v;
}
// A 16x32 f16: lane<16 -> row=lane, K in {0..7}(v0-3) and {16..23}(v4-7);
//              lane>=16 -> row=lane-16, K in {8..15} and {24..31}.
__device__ __forceinline__ v16h load_a_frag(const _Float16* base, int ld, int lane) {
  const int half = lane >> 4, m = lane & 15, kb = half * 8;
  const _Float16* rp = base + (size_t)m * ld;
  return pack2(*(const v8h*)(rp + kb), *(const v8h*)(rp + 16 + kb));
}
// B 32x16, same K order as A (K in {8h..8h+7} and {16+8h..16+8h+7}): source tile stored n-major [n][k] (stride ld).
__device__ __forceinline__ v16h load_bn_frag(const _Float16* base, int ld, int lane) {
  const int half = lane >> 4, n = lane & 15;
  const _Float16* rp = base + (size_t)n * ld + half * 8;
  return pack2(*(const v8h*)(rp), *(const v8h*)(rp + 16));
}

// ---------------- weight convert + transpose: d[n][k] = (f16) s[k][n] ----------------
__global__ __launch_bounds__(256)
void convT_kernel(const float* __restrict__ s, _Float16* __restrict__ d, int Kd, int N) {
  __shared__ _Float16 tile[32][33];
  const int kt = blockIdx.x * 32, nt = blockIdx.y * 32;
  const int tx = threadIdx.x & 31, ty = threadIdx.x >> 5;   // 32 x 8
#pragma unroll
  for (int j = 0; j < 32; j += 8)
    tile[ty + j][tx] = (_Float16)s[(size_t)(kt + ty + j) * N + nt + tx];
  __syncthreads();
#pragma unroll
  for (int j = 0; j < 32; j += 8)
    d[(size_t)(nt + ty + j) * Kd + kt + tx] = tile[tx][ty + j];
}

// ---------------- LayerNorm (vectorized) ----------------
__global__ __launch_bounds__(256)
void ln_kernel(const float* __restrict__ x, const float* __restrict__ g,
               const float* __restrict__ bta, float* __restrict__ outF,
               _Float16* __restrict__ outH) {
  const int row = blockIdx.x, tid = threadIdx.x;
  const float* xr = x + (size_t)row * cD;
  __shared__ float red[256];
  const float4 v4 = *(const float4*)(xr + tid * 4);
  red[tid] = v4.x + v4.y + v4.z + v4.w; __syncthreads();
  for (int s2 = 128; s2 > 0; s2 >>= 1) { if (tid < s2) red[tid] += red[tid + s2]; __syncthreads(); }
  float mu = red[0] * (1.f / cD); __syncthreads();
  float d0 = v4.x - mu, d1 = v4.y - mu, d2 = v4.z - mu, d3 = v4.w - mu;
  red[tid] = d0*d0 + d1*d1 + d2*d2 + d3*d3; __syncthreads();
  for (int s2 = 128; s2 > 0; s2 >>= 1) { if (tid < s2) red[tid] += red[tid + s2]; __syncthreads(); }
  const float rstd = rsqrtf(red[0] * (1.f / cD) + 1e-5f);
  const float4 g4 = *(const float4*)(g + tid * 4);
  const float4 b4 = *(const float4*)(bta + tid * 4);
  float y0 = d0*rstd*g4.x + b4.x, y1 = d1*rstd*g4.y + b4.y;
  float y2 = d2*rstd*g4.z + b4.z, y3 = d3*rstd*g4.w + b4.w;
  if (outF) { float4 o = {y0, y1, y2, y3}; *(float4*)(outF + (size_t)row*cD + tid*4) = o; }
  if (outH) {
    v4h o; o[0] = (_Float16)y0; o[1] = (_Float16)y1; o[2] = (_Float16)y2; o[3] = (_Float16)y3;
    *(v4h*)(outH + (size_t)row*cD + tid*4) = o;
  }
}

// ---------------- selection: entity pooling + hard top-1 rule / entity (one block per batch) ----------------
__global__ __launch_bounds__(256)
void select_kernel(const float* __restrict__ h, const float* __restrict__ rules,
                   const float* __restrict__ W_ent, const float* __restrict__ b_ent,
                   const float* __restrict__ Wq_er, const float* __restrict__ Wk_er,
                   const float* __restrict__ Wq_es, const float* __restrict__ Wk_es,
                   float* __restrict__ egate_g, float* __restrict__ rulew_g) {
  const int b = blockIdx.x, tid = threadIdx.x;
  __shared__ float pooled[cE][cD];
  __shared__ float ent[cE][cSD];
  __shared__ float qer[cKT][cR][cSD];
  __shared__ float ker[cE][cSD];
  __shared__ float ser[cKT][cR][cE];
  __shared__ int   selr[cKT];
  __shared__ float qes[cKT][16];
  __shared__ float kes[cE][16];
  __shared__ float ses[cKT][cE];
  __shared__ int   sele[cKT];
  const int seg0[3] = {0, 128, 192};
  const int slen[3] = {128, 64, 512};
  for (int e = 0; e < cE; ++e)
    for (int d = tid; d < cD; d += 256) {
      float s = 0.f;
      const float* hp = h + ((size_t)(b*cT + seg0[e]))*cD + d;
      for (int t = 0; t < slen[e]; ++t) s += hp[(size_t)t*cD];
      pooled[e][d] = s / (float)slen[e];
    }
  __syncthreads();
  if (tid < cE*cSD) {
    int e = tid / cSD, si = tid % cSD;
    float s = b_ent[si];
    for (int d = 0; d < cD; ++d) s += pooled[e][d] * W_ent[d*cSD + si];
    ent[e][si] = s;
  }
  __syncthreads();
  for (int j = tid; j < cKT*cR*cSD; j += 256) {
    int k = j >> 9, rest = j & 511, rr = rest >> 5, si = rest & 31;
    float s = 0.f;
    const float* re = rules + ((size_t)(k*cR + rr))*cHD;
    for (int hh = 0; hh < cHD; ++hh) s += re[hh] * Wq_er[hh*cSD + si];
    qer[k][rr][si] = s;
  }
  if (tid < cE*cSD) {
    int e = tid / cSD, si = tid % cSD;
    float s = 0.f;
    for (int d2 = 0; d2 < cSD; ++d2) s += ent[e][d2] * Wk_er[d2*cSD + si];
    ker[e][si] = s;
  }
  __syncthreads();
  if (tid < cKT*cR*cE) {
    int k = tid / (cR*cE), rest = tid % (cR*cE), rr = rest / cE, e = rest % cE;
    float s = 0.f;
    for (int d2 = 0; d2 < cSD; ++d2) s += qer[k][rr][d2] * ker[e][d2];
    ser[k][rr][e] = s * 0.17677669529663687f;   // 1/sqrt(32)
  }
  __syncthreads();
  if (tid < cKT) {                               // hard top-1 over (r,e), first-max on ties
    float best = -1e30f; int bi = 0;
    for (int idx = 0; idx < cR*cE; ++idx) {
      float v = ser[tid][idx/cE][idx%cE];
      if (v > best) { best = v; bi = idx; }
    }
    selr[tid] = bi / cE;
  }
  __syncthreads();
  if (tid < cKT*16) {
    int k = tid / 16, i = tid % 16;
    const float* re = rules + ((size_t)(k*cR + selr[k]))*cHD;
    float s = 0.f;
    for (int hh = 0; hh < cHD; ++hh) s += re[hh] * Wq_es[((size_t)(k*cHD + hh))*16 + i];
    qes[k][i] = s;
  }
  if (tid < cE*16) {
    int e = tid / 16, i = tid % 16;
    float s = 0.f;
    for (int d2 = 0; d2 < cSD; ++d2) s += ent[e][d2] * Wk_es[d2*16 + i];
    kes[e][i] = s;
  }
  __syncthreads();
  if (tid < cKT*cE) {
    int k = tid / cE, e = tid % cE;
    float s = 0.f;
    for (int i = 0; i < 16; ++i) s += qes[k][i] * kes[e][i];
    ses[k][e] = s * 0.25f;                       // 1/sqrt(16)
  }
  __syncthreads();
  if (tid < cKT) {
    float best = -1e30f; int bi = 0;
    for (int e = 0; e < cE; ++e) { float v = ses[tid][e]; if (v > best) { best = v; bi = e; } }
    sele[tid] = bi;
  }
  __syncthreads();
  if (tid == 0) {
    float eg[4] = {0.f, 0.f, 0.f, 0.f};
    float rw[cR];
    for (int r2 = 0; r2 < cR; ++r2) rw[r2] = 0.f;
    for (int k = 0; k < cKT; ++k) { eg[sele[k]] += 1.f; rw[selr[k]] += 1.f; }
    for (int e = 0; e < 4; ++e) egate_g[b*4 + e] = eg[e];
    for (int r2 = 0; r2 < cR; ++r2) rulew_g[b*cR + r2] = rw[r2];
  }
}

__global__ void gate_kernel(const float* __restrict__ h, const float* __restrict__ egate,
                            _Float16* __restrict__ hg) {
  size_t u = (size_t)blockIdx.x * 256 + threadIdx.x;   // units of 8 elements
  if (u >= (size_t)BT * cD / 8) return;
  size_t i = u * 8;
  size_t bt = i >> 10;
  int b = (int)(bt / cT), t = (int)(bt % cT);
  int e = t < 128 ? 0 : (t < 192 ? 1 : 2);
  const float gv = egate[b*4 + e];
  const float4 a = *(const float4*)(h + i);
  const float4 c = *(const float4*)(h + i + 4);
  v8h o;
  o[0] = (_Float16)(a.x*gv); o[1] = (_Float16)(a.y*gv);
  o[2] = (_Float16)(a.z*gv); o[3] = (_Float16)(a.w*gv);
  o[4] = (_Float16)(c.x*gv); o[5] = (_Float16)(c.y*gv);
  o[6] = (_Float16)(c.z*gv); o[7] = (_Float16)(c.w*gv);
  *(v8h*)(hg + i) = o;
}

// ---------------- generic WMMA GEMM, double-buffered async staging ----------------
// C = act(scale*A*Bt^T + biasMul*bias) (+ residual)
// A: [M,Kd] f16 row-major; Bt: [N,Kd] f16 row-major (pre-transposed). M%64==0, N%64==0, Kd%32==0.
__global__ __launch_bounds__(256)
void gemm_f16_kernel(const _Float16* __restrict__ A, const _Float16* __restrict__ Bt,
                     const float* __restrict__ bias, float biasMul,
                     const float* __restrict__ residual, float scale, int act,
                     int M, int N, int Kd,
                     _Float16* __restrict__ outH, float* __restrict__ outF) {
  __shared__ __align__(16) _Float16 As[2][64][40];
  __shared__ __align__(16) _Float16 BsT[2][64][40];
  const int tid = threadIdx.x, wave = tid >> 5, lane = tid & 31;
  const int half = lane >> 4, nlow = lane & 15;
  const int m0 = (wave & 3) * 16, n0w = (wave >> 2) * 32;
  const int blockM = blockIdx.x * 64, blockN = blockIdx.y * 64;
  const int srow = tid >> 2, scol = (tid & 3) * 8;     // one v8h per thread per tile
  const _Float16* Aptr = A  + (size_t)(blockM + srow) * Kd + scol;
  const _Float16* Bptr = Bt + (size_t)(blockN + srow) * Kd + scol;
  v8f c0 = {}, c1 = {};
  // prologue: stage first K tile into buffer 0
  async_g2l_b128((const v8h*)Aptr, (v8h*)&As[0][srow][scol]);
  async_g2l_b128((const v8h*)Bptr, (v8h*)&BsT[0][srow][scol]);
  int ib = 0;
  for (int kt = 0; kt < Kd; kt += 32, ib ^= 1) {
    if (kt + 32 < Kd) {   // issue next tile into the other buffer, retire only the current one
      async_g2l_b128((const v8h*)(Aptr + kt + 32), (v8h*)&As[ib ^ 1][srow][scol]);
      async_g2l_b128((const v8h*)(Bptr + kt + 32), (v8h*)&BsT[ib ^ 1][srow][scol]);
      if (kt + 64 < Kd) {
        __builtin_prefetch(Aptr + kt + 64, 0, 0);
        __builtin_prefetch(Bptr + kt + 64, 0, 0);
      }
      wait_async<2>();
    } else {
      wait_async<0>();
    }
    __syncthreads();
    v16h a  = load_a_frag(&As[ib][m0][0], 40, lane);
    v16h b0 = load_bn_frag(&BsT[ib][n0w][0], 40, lane);
    v16h b1 = load_bn_frag(&BsT[ib][n0w + 16][0], 40, lane);
    c0 = __builtin_amdgcn_wmma_f32_16x16x32_f16(false, a, false, b0, (short)0, c0, false, false);
    c1 = __builtin_amdgcn_wmma_f32_16x16x32_f16(false, a, false, b1, (short)0, c1, false, false);
    __syncthreads();
  }
#pragma unroll
  for (int v = 0; v < 8; ++v) {
    int gm = blockM + m0 + v + 8*half;
    int gn0 = blockN + n0w + nlow, gn1 = gn0 + 16;
    if (gm >= M) continue;
    float x0 = c0[v]*scale + (bias ? bias[gn0]*biasMul : 0.f);
    float x1 = c1[v]*scale + (bias ? bias[gn1]*biasMul : 0.f);
    if (act == 1) { x0 = fmaxf(x0, 0.f); x1 = fmaxf(x1, 0.f); }
    else if (act == 2) {
      x0 = 0.5f*x0*(1.f + erff(x0*0.70710678118654752f));
      x1 = 0.5f*x1*(1.f + erff(x1*0.70710678118654752f));
    }
    if (residual) { x0 += residual[(size_t)gm*N + gn0]; x1 += residual[(size_t)gm*N + gn1]; }
    if (outH) { outH[(size_t)gm*N + gn0] = (_Float16)x0; outH[(size_t)gm*N + gn1] = (_Float16)x1; }
    if (outF) { outF[(size_t)gm*N + gn0] = x0;           outF[(size_t)gm*N + gn1] = x1; }
  }
}

// ---------------- WMMA flash attention per (q-tile, head, batch); output pre-weighted by rulew ----------------
__global__ __launch_bounds__(256)
void attn_kernel(const _Float16* __restrict__ Qg, const _Float16* __restrict__ Kg,
                 const _Float16* __restrict__ Vg, const float* __restrict__ amask,
                 const float* __restrict__ rulew, _Float16* __restrict__ wao) {
  const int qt = blockIdx.x, rr = blockIdx.y, b = blockIdx.z;
  const int tid = threadIdx.x, wave = tid >> 5, lane = tid & 31;
  const int half = lane >> 4, nlow = lane & 15;
  const int msub = wave & 3, grp = wave >> 2;
  __shared__ __align__(16) _Float16 Qs[64][72];
  __shared__ __align__(16) _Float16 Ks[2][64][72];   // double-buffered K tiles
  __shared__ __align__(16) _Float16 VsT[64][72];     // transposed: [h][key]
  __shared__ __align__(16) _Float16 Ps[64][72];
  __shared__ __align__(16) float Ss[64][68];
  __shared__ float mrow_s[64];
  __shared__ float lrow_s[64];

  // --- Q tile: Tensor Data Mover 2D tile load (64x64 f16 with 16B/row LDS pad) when available ---
#if __has_builtin(__builtin_amdgcn_tensor_load_to_lds) && __has_builtin(__builtin_amdgcn_s_wait_tensorcnt)
  if (wave == 0) {
    const _Float16* gq = &Qg[((size_t)(b*cT + qt*64))*cD + rr*cHD];
    unsigned long long ga = (unsigned long long)(size_t)gq;
    u32x4 g0;
    g0[0] = 1u;                                        // count=1 valid descriptor
    g0[1] = (unsigned)(size_t)&Qs[0][0];               // lds_addr
    g0[2] = (unsigned)ga;                              // global_addr[31:0]
    g0[3] = (unsigned)((ga >> 32) & 0x1FFFFFFu) | (2u << 30);  // global_addr[56:32], type=2
    i32x8 g1;
    g1[0] = (1 << 16)                                  // data_size = 2 bytes
          | (1 << 20) | (4 << 22) | (3 << 25);         // pad: every 32 DW add 4 DW (64->72 halves)
    g1[1] = (int)((cD & 0xFFFFu) << 16);               // tensor_dim0 lo16 -> bits 63:48
    g1[2] = (int)((cD >> 16) | ((cT & 0xFFFFu) << 16));// tensor_dim0 hi, tensor_dim1 lo
    g1[3] = (int)((cT >> 16) | (64u << 16));           // tensor_dim1 hi, tile_dim0 = 64
    g1[4] = 64;                                        // tile_dim1 = 64
    g1[5] = cD;                                        // tensor_dim0_stride lo
    g1[6] = 0; g1[7] = 0;
    i32x4 gz = {0, 0, 0, 0};
#if __clang_major__ >= 23
    i32x8 gz8 = {0, 0, 0, 0, 0, 0, 0, 0};
    __builtin_amdgcn_tensor_load_to_lds(g0, g1, gz, gz, gz8, 0);
#else
    __builtin_amdgcn_tensor_load_to_lds(g0, g1, gz, gz, 0);
#endif
    __builtin_amdgcn_s_wait_tensorcnt(0);
  }
#else
#pragma unroll
  for (int u = tid; u < 64*8; u += 256) {            // 8-half chunks
    int r0 = u >> 3, c8 = (u & 7) * 8;
    *(v8h*)&Qs[r0][c8] = *(const v8h*)&Qg[((size_t)(b*cT + qt*64 + r0))*cD + rr*cHD + c8];
  }
#endif
  if (tid < 64) { mrow_s[tid] = -1e30f; lrow_s[tid] = 0.f; }

  // pass 1: row maxima of (QK^T + mask), double-buffered K staging
  {
#pragma unroll
    for (int u = tid; u < 64*8; u += 256) {
      int r0 = u >> 3, c8 = (u & 7) * 8;
      async_g2l_b128((const v8h*)&Kg[((size_t)(b*cT + r0))*cD + rr*cHD + c8],
                     (v8h*)&Ks[0][r0][c8]);
    }
  }
  int ib = 0;
  for (int kt = 0; kt < NKT; ++kt, ib ^= 1) {
    if (kt + 1 < NKT) {
#pragma unroll
      for (int u = tid; u < 64*8; u += 256) {
        int r0 = u >> 3, c8 = (u & 7) * 8;
        async_g2l_b128((const v8h*)&Kg[((size_t)(b*cT + (kt+1)*64 + r0))*cD + rr*cHD + c8],
                       (v8h*)&Ks[ib ^ 1][r0][c8]);
      }
      wait_async<2>();
    } else {
      wait_async<0>();
    }
    __syncthreads();
#pragma unroll
    for (int j = 0; j < 2; ++j) {
      int nsub = grp*2 + j;
      v8f acc = {};
#pragma unroll
      for (int kc = 0; kc < cHD; kc += 32) {
        v16h a  = load_a_frag(&Qs[msub*16][kc], 72, lane);
        v16h bf = load_bn_frag(&Ks[ib][nsub*16][kc], 72, lane);   // B[k][n] = K[n][k]
        acc = __builtin_amdgcn_wmma_f32_16x16x32_f16(false, a, false, bf, (short)0, acc, false, false);
      }
#pragma unroll
      for (int v = 0; v < 8; ++v) Ss[msub*16 + v + 8*half][nsub*16 + nlow] = acc[v];
    }
    __syncthreads();
    if (tid < 64) {
      const float* mp = amask + ((size_t)(b*cT + qt*64 + tid))*cT + kt*64;
      float mx = mrow_s[tid];
#pragma unroll
      for (int j4 = 0; j4 < 64; j4 += 4) {
        const float4 mv = *(const float4*)(mp + j4);
        mx = fmaxf(mx, Ss[tid][j4+0] + mv.x);
        mx = fmaxf(mx, Ss[tid][j4+1] + mv.y);
        mx = fmaxf(mx, Ss[tid][j4+2] + mv.z);
        mx = fmaxf(mx, Ss[tid][j4+3] + mv.w);
      }
      mrow_s[tid] = mx;
    }
    __syncthreads();
  }

  // pass 2: P = exp(S+mask-m); O += P*V (single-buffered: V transpose copy serializes staging)
  v8f o0 = {}, o1 = {};
  const int hbase = grp * 32;
  for (int kt = 0; kt < NKT; ++kt) {
#pragma unroll
    for (int u = tid; u < 64*8; u += 256) {
      int r0 = u >> 3, c8 = (u & 7) * 8;
      size_t src = ((size_t)(b*cT + kt*64 + r0))*cD + rr*cHD + c8;
      async_g2l_b128((const v8h*)&Kg[src], (v8h*)&Ks[0][r0][c8]);
      v8h vv = *(const v8h*)&Vg[src];
#pragma unroll
      for (int jj = 0; jj < 8; ++jj) VsT[c8 + jj][r0] = vv[jj];   // transpose V into [h][key]
    }
    wait_async<0>();
    __syncthreads();
#pragma unroll
    for (int j = 0; j < 2; ++j) {
      int nsub = grp*2 + j;
      v8f acc = {};
#pragma unroll
      for (int kc = 0; kc < cHD; kc += 32) {
        v16h a  = load_a_frag(&Qs[msub*16][kc], 72, lane);
        v16h bf = load_bn_frag(&Ks[0][nsub*16][kc], 72, lane);
        acc = __builtin_amdgcn_wmma_f32_16x16x32_f16(false, a, false, bf, (short)0, acc, false, false);
      }
#pragma unroll
      for (int v = 0; v < 8; ++v) Ss[msub*16 + v + 8*half][nsub*16 + nlow] = acc[v];
    }
    __syncthreads();
    if (tid < 64) {
      const float* mp = amask + ((size_t)(b*cT + qt*64 + tid))*cT + kt*64;
      float mx = mrow_s[tid], s = 0.f;
#pragma unroll
      for (int j8 = 0; j8 < 64; j8 += 8) {
        const float4 ma = *(const float4*)(mp + j8);
        const float4 mb = *(const float4*)(mp + j8 + 4);
        v8h p8; float pv;
        pv = __expf(Ss[tid][j8+0] + ma.x - mx); p8[0] = (_Float16)pv; s += pv;
        pv = __expf(Ss[tid][j8+1] + ma.y - mx); p8[1] = (_Float16)pv; s += pv;
        pv = __expf(Ss[tid][j8+2] + ma.z - mx); p8[2] = (_Float16)pv; s += pv;
        pv = __expf(Ss[tid][j8+3] + ma.w - mx); p8[3] = (_Float16)pv; s += pv;
        pv = __expf(Ss[tid][j8+4] + mb.x - mx); p8[4] = (_Float16)pv; s += pv;
        pv = __expf(Ss[tid][j8+5] + mb.y - mx); p8[5] = (_Float16)pv; s += pv;
        pv = __expf(Ss[tid][j8+6] + mb.z - mx); p8[6] = (_Float16)pv; s += pv;
        pv = __expf(Ss[tid][j8+7] + mb.w - mx); p8[7] = (_Float16)pv; s += pv;
        *(v8h*)&Ps[tid][j8] = p8;
      }
      lrow_s[tid] += s;
    }
    __syncthreads();
#pragma unroll
    for (int kc = 0; kc < 64; kc += 32) {
      v16h a  = load_a_frag(&Ps[msub*16][kc], 72, lane);
      v16h b0 = load_bn_frag(&VsT[hbase][kc], 72, lane);        // B[k][h] = VsT[h][k]
      v16h b1 = load_bn_frag(&VsT[hbase + 16][kc], 72, lane);
      o0 = __builtin_amdgcn_wmma_f32_16x16x32_f16(false, a, false, b0, (short)0, o0, false, false);
      o1 = __builtin_amdgcn_wmma_f32_16x16x32_f16(false, a, false, b1, (short)0, o1, false, false);
    }
    __syncthreads();
  }
  const float rwv = rulew[b*cR + rr];
#pragma unroll
  for (int v = 0; v < 8; ++v) {
    int rl = msub*16 + v + 8*half;
    float inv = rwv / lrow_s[rl];
    size_t base = (((size_t)(b*cR + rr))*cT + qt*64 + rl) * cHD;
    wao[base + hbase + nlow]      = (_Float16)(o0[v]*inv);
    wao[base + hbase + 16 + nlow] = (_Float16)(o1[v]*inv);
  }
}

__global__ void reduce_wao(const _Float16* __restrict__ wao, _Float16* __restrict__ selh) {
  int u = blockIdx.x * 256 + threadIdx.x;          // units of 8 halves
  if (u >= cB*cT*cHD/8) return;
  int h8 = (u & 7) * 8, bt = u >> 3;
  int b = bt / cT, t = bt % cT;
  float acc[8] = {0.f,0.f,0.f,0.f,0.f,0.f,0.f,0.f};
  for (int r = 0; r < cR; ++r) {
    v8h w = *(const v8h*)&wao[(((size_t)(b*cR + r))*cT + t)*cHD + h8];
#pragma unroll
    for (int j = 0; j < 8; ++j) acc[j] += (float)w[j];
  }
  v8h o;
#pragma unroll
  for (int j = 0; j < 8; ++j) o[j] = (_Float16)acc[j];
  *(v8h*)&selh[(size_t)u * 8] = o;
}

// ---------------- host launch ----------------
extern "C" void kernel_launch(void* const* d_in, const int* in_sizes, int n_in,
                              void* d_out, int out_size, void* d_ws, size_t ws_size,
                              hipStream_t stream) {
  (void)in_sizes; (void)n_in; (void)out_size; (void)ws_size;
  const float* hidden = (const float*)d_in[0];
  const float* amask  = (const float*)d_in[1];
  const float* rules  = (const float*)d_in[2];
  const float* W_ent  = (const float*)d_in[3];
  const float* b_ent  = (const float*)d_in[4];
  const float* Wq_er  = (const float*)d_in[5];
  const float* Wk_er  = (const float*)d_in[6];
  const float* Wq_es  = (const float*)d_in[7];
  const float* Wk_es  = (const float*)d_in[8];
  const float* Wq     = (const float*)d_in[9];
  const float* bq     = (const float*)d_in[10];
  const float* Wk     = (const float*)d_in[11];
  const float* bk     = (const float*)d_in[12];
  const float* Wv     = (const float*)d_in[13];
  const float* bv     = (const float*)d_in[14];
  const float* Wo     = (const float*)d_in[15];
  const float* bo     = (const float*)d_in[16];
  const float* ln1_g  = (const float*)d_in[17];
  const float* ln1_b  = (const float*)d_in[18];
  const float* ln2_g  = (const float*)d_in[19];
  const float* ln2_b  = (const float*)d_in[20];
  const float* f11    = (const float*)d_in[21];
  const float* f12    = (const float*)d_in[22];
  const float* f21    = (const float*)d_in[23];
  const float* f22    = (const float*)d_in[24];

  char* p = (char*)d_ws;
  auto alloc = [&](size_t bytes) -> char* {
    char* r = p; p += (bytes + 255) & ~(size_t)255; return r;
  };
  float*    hbuf  = (float*)   alloc((size_t)BT*cD*4);
  float*    egate = (float*)   alloc(cB*4*4);
  float*    rwg   = (float*)   alloc(cB*cR*4);
  _Float16* hg16  = (_Float16*)alloc((size_t)BT*cD*2);
  _Float16* wqT   = (_Float16*)alloc((size_t)cD*cD*2);
  _Float16* wkT   = (_Float16*)alloc((size_t)cD*cD*2);
  _Float16* wvT   = (_Float16*)alloc((size_t)cD*cD*2);
  _Float16* woT   = (_Float16*)alloc((size_t)cHD*cD*2);
  _Float16* w11T  = (_Float16*)alloc((size_t)cD*(cD/2)*2);
  _Float16* w12T  = (_Float16*)alloc((size_t)(cD/2)*(4*cD)*2);
  _Float16* w21T  = (_Float16*)alloc((size_t)(4*cD)*(cD/2)*2);
  _Float16* w22T  = (_Float16*)alloc((size_t)(cD/2)*cD*2);
  _Float16* q16   = (_Float16*)alloc((size_t)BT*cD*2);
  _Float16* k16   = (_Float16*)alloc((size_t)BT*cD*2);
  _Float16* vv16  = (_Float16*)alloc((size_t)BT*cD*2);
  _Float16* wao16 = (_Float16*)alloc((size_t)cB*cR*cT*cHD*2);
  _Float16* selh  = (_Float16*)alloc((size_t)BT*cHD*2);
  float*    h2    = (float*)   alloc((size_t)BT*cD*4);
  _Float16* x16   = (_Float16*)alloc((size_t)BT*cD*2);
  _Float16* t1    = (_Float16*)alloc((size_t)BT*(cD/2)*2);
  _Float16* t2    = (_Float16*)alloc((size_t)BT*(4*cD)*2);
  _Float16* t3    = (_Float16*)alloc((size_t)BT*(cD/2)*2);

  // weight convert + transpose: dst[n][k] = (f16) src[k][n]
  auto convT = [&](const float* s, _Float16* d, int Kd, int N) {
    dim3 g(Kd/32, N/32);
    convT_kernel<<<g, 256, 0, stream>>>(s, d, Kd, N);
  };
  convT(Wq,  wqT,  cD,   cD);
  convT(Wk,  wkT,  cD,   cD);
  convT(Wv,  wvT,  cD,   cD);
  convT(Wo,  woT,  cHD,  cD);
  convT(f11, w11T, cD,   cD/2);
  convT(f12, w12T, cD/2, 4*cD);
  convT(f21, w21T, 4*cD, cD/2);
  convT(f22, w22T, cD/2, cD);

  ln_kernel<<<BT, 256, 0, stream>>>(hidden, ln1_g, ln1_b, hbuf, nullptr);
  select_kernel<<<cB, 256, 0, stream>>>(hbuf, rules, W_ent, b_ent,
                                        Wq_er, Wk_er, Wq_es, Wk_es, egate, rwg);
  gate_kernel<<<((size_t)BT*cD/8 + 255)/256, 256, 0, stream>>>(hbuf, egate, hg16);

  auto gemm = [&](const _Float16* A, const _Float16* Bt, const float* bias, float biasMul,
                  const float* residual, float scale, int act, int M, int N, int Kd,
                  _Float16* oh, float* of) {
    dim3 g((M + 63)/64, N/64);
    gemm_f16_kernel<<<g, 256, 0, stream>>>(A, Bt, bias, biasMul, residual, scale, act,
                                           M, N, Kd, oh, of);
  };
  const float qscale = 0.125f;   // HD^-0.5
  gemm(hg16, wqT, bq, qscale, nullptr, qscale, 0, BT, cD, cD, q16, nullptr);
  gemm(hg16, wkT, bk, 1.f,    nullptr, 1.f,    0, BT, cD, cD, k16, nullptr);
  gemm(hg16, wvT, bv, 1.f,    nullptr, 1.f,    0, BT, cD, cD, vv16, nullptr);

  dim3 ga(NKT, cR, cB);
  attn_kernel<<<ga, 256, 0, stream>>>(q16, k16, vv16, amask, rwg, wao16);
  reduce_wao<<<(cB*cT*cHD/8 + 255)/256, 256, 0, stream>>>(wao16, selh);

  // attn_out = selsum @ Wo + K*bo ; h2 = hidden + attn_out
  gemm(selh, woT, bo, (float)cKT, hidden, 1.f, 0, BT, cD, cHD, nullptr, h2);

  ln_kernel<<<BT, 256, 0, stream>>>(h2, ln2_g, ln2_b, nullptr, x16);

  gemm(x16, w11T, nullptr, 0.f, nullptr, 1.f, 1, BT, cD/2, cD,   t1, nullptr);  // relu
  gemm(t1,  w12T, nullptr, 0.f, nullptr, 1.f, 2, BT, 4*cD, cD/2, t2, nullptr);  // gelu (exact)
  gemm(t2,  w21T, nullptr, 0.f, nullptr, 1.f, 1, BT, cD/2, 4*cD, t3, nullptr);  // relu
  gemm(t3,  w22T, nullptr, 0.f, h2,      1.f, 0, BT, cD,   cD/2, nullptr, (float*)d_out);
}
